// MMBERTLayer_11819749999068
// MI455X (gfx1250) — compile-verified, matrix operand order checked by reading, not run on hardware
//
#include <hip/hip_runtime.h>
#include <hip/hip_bf16.h>

// ---------------------------------------------------------------------------
// MMBERT layer for gfx1250 (MI455X). All GEMMs run on v_wmma_f32_16x16x32_f16
// (wave32 WMMA), fp32 accumulation. GEMMs are uniformly "NT": A is [M,K]
// row-major f16, B is stored [N,K] row-major f16 (weights pre-transposed at
// conversion; attention K-matrix is naturally NT; V is transposed per head).
// Global->LDS staging uses GLOBAL_LOAD_ASYNC_TO_LDS_B128 when the toolchain
// declares the builtin (ASYNCcnt path), else vector load + ds_store.
// ---------------------------------------------------------------------------

typedef __attribute__((ext_vector_type(16))) _Float16     v16h;
typedef __attribute__((ext_vector_type(8)))  _Float16     v8h;
typedef __attribute__((ext_vector_type(8)))  float        v8f;
typedef __attribute__((ext_vector_type(4)))  unsigned int v4u;
typedef __attribute__((ext_vector_type(4)))  int          v4i;

#define HIDd 768
#define NHd  12
#define DHd  64
#define FFd  3072
#define Bd   8
#define Sd   512
#define Md   (Bd * Sd)          // 4096 rows
#define ZBH  (Bd * NHd)         // 96 attention batches
#define LST  40                 // LDS row stride in halves (80B, 16B aligned, conflict-free)

#if __has_builtin(__builtin_amdgcn_global_load_async_to_lds_b128)
#define USE_ASYNC 1
#else
#define USE_ASYNC 0
#endif

#if USE_ASYNC
__device__ __forceinline__ void async_b128(const _Float16* g, _Float16* l) {
    // generic->AS1: same 64-bit value; generic LDS pointer low 32 bits == LDS offset.
    __builtin_amdgcn_global_load_async_to_lds_b128(
        (__attribute__((address_space(1))) v4i*)(unsigned long long)g,
        (__attribute__((address_space(3))) v4i*)(unsigned int)(unsigned long long)l,
        0, 0);
}
__device__ __forceinline__ void wait_async0() {
#if __has_builtin(__builtin_amdgcn_s_wait_asynccnt)
    __builtin_amdgcn_s_wait_asynccnt(0);
#else
    asm volatile("s_wait_asynccnt 0" ::: "memory");
#endif
}
#endif

// ---------------------------------------------------------------------------
// Generic batched NT GEMM:  C[z][M,N](f32/f16) = alpha * A[z] @ Bt[z]^T + bias
// Template tile: BM x BN x 32, WR x WC waves (wave32), each wave owns a
// (BM/WR) x (BN/WC) output tile (MI x NI WMMA frags), all B-frags loaded
// before the WMMA group so the matrix pipe issues back-to-back.
// Fragment layouts per ISA 7.12.2:
//   A 16x32 f16: half=lane>>4, m=lane&15; 16 halves = two contiguous 16B runs
//     at k-offsets {8*half, 16+8*half}. B stored [n][k] in LDS, identical.
//   C/D: VGPR r -> row M = r + 8*half, col N = lane&15.
// ---------------------------------------------------------------------------
template <int BM, int BN, int WR, int WC>
__global__ __launch_bounds__(WR * WC * 32) void k_gemm_nt(
    const _Float16* __restrict__ A, const _Float16* __restrict__ Bt,
    float* __restrict__ C, _Float16* __restrict__ Ch,
    const float* __restrict__ bias, float alpha,
    int M, int N, int K, int lda, int ldb, int ldc, int nh,
    long aB, long aH, long bB, long bH, long cB, long cH)
{
    constexpr int NT = WR * WC * 32;
    constexpr int MI = BM / WR / 16;   // A frags per wave
    constexpr int NI = BN / WC / 16;   // B frags per wave
    static_assert(NT == 2 * BM && BM == BN, "load mapping assumes NT==2*BM, BM==BN");

    __shared__ _Float16 sA[BM * LST];
    __shared__ _Float16 sB[BN * LST];

    const int z  = blockIdx.z;
    const int bz = z / nh;
    const int hz = z - bz * nh;
    A  += (long)bz * aB + (long)hz * aH;
    Bt += (long)bz * bB + (long)hz * bH;
    const long coff = (long)bz * cB + (long)hz * cH;

    const int m0 = blockIdx.x * BM;
    const int n0 = blockIdx.y * BN;
    const int tid  = threadIdx.x;
    const int w    = tid >> 5;
    const int lane = tid & 31;
    const int half = lane >> 4;
    const int l16  = lane & 15;
    const int wr   = w / WC;           // wave row
    const int wc   = w - wr * WC;      // wave col

    const int rload = tid >> 1;          // 0..BM-1 tile row
    const int cload = (tid & 1) * 16;    // 0 or 16 within 32-wide k slab

    v8f acc[MI][NI] = {};

    for (int k0 = 0; k0 < K; k0 += 32) {
        const _Float16* ga = A  + (size_t)(m0 + rload) * lda + k0 + cload;
        const _Float16* gb = Bt + (size_t)(n0 + rload) * ldb + k0 + cload;
        _Float16* la = &sA[rload * LST + cload];
        _Float16* lb = &sB[rload * LST + cload];
#if USE_ASYNC
        async_b128(ga,     la);
        async_b128(ga + 8, la + 8);
        async_b128(gb,     lb);
        async_b128(gb + 8, lb + 8);
        wait_async0();                 // own-wave async writes landed in LDS
        __syncthreads();               // all waves' writes visible
#else
        v8h a0 = ((const v8h*)ga)[0], a1 = ((const v8h*)ga)[1];
        v8h b0 = ((const v8h*)gb)[0], b1 = ((const v8h*)gb)[1];
        *(v8h*)la       = a0;
        *(v8h*)(la + 8) = a1;
        *(v8h*)lb       = b0;
        *(v8h*)(lb + 8) = b1;
        __syncthreads();
#endif

        union FR { v16h h; v4u u[2]; };
        FR fa[MI], fb[NI];
#pragma unroll
        for (int mi = 0; mi < MI; ++mi) {
            const int am = wr * (MI * 16) + mi * 16 + l16;
            fa[mi].u[0] = *(const v4u*)&sA[am * LST + 8 * half];
            fa[mi].u[1] = *(const v4u*)&sA[am * LST + 16 + 8 * half];
        }
#pragma unroll
        for (int ni = 0; ni < NI; ++ni) {
            const int bn = wc * (NI * 16) + ni * 16 + l16;
            fb[ni].u[0] = *(const v4u*)&sB[bn * LST + 8 * half];
            fb[ni].u[1] = *(const v4u*)&sB[bn * LST + 16 + 8 * half];
        }
#pragma unroll
        for (int mi = 0; mi < MI; ++mi)
#pragma unroll
            for (int ni = 0; ni < NI; ++ni)
                acc[mi][ni] = __builtin_amdgcn_wmma_f32_16x16x32_f16(
                    false, fa[mi].h, false, fb[ni].h, (short)0, acc[mi][ni], false, false);
        __syncthreads();
    }

    // ---- epilogue: scale + bias, write f32 and/or f16
#pragma unroll
    for (int mi = 0; mi < MI; ++mi) {
#pragma unroll
        for (int ni = 0; ni < NI; ++ni) {
            const int gn = n0 + wc * (NI * 16) + ni * 16 + l16;
            const float bv = bias ? bias[gn] : 0.0f;
#pragma unroll
            for (int r = 0; r < 8; ++r) {
                const int gm = m0 + wr * (MI * 16) + mi * 16 + r + 8 * half;
                const float val = acc[mi][ni][r] * alpha + bv;
                const size_t o = (size_t)coff + (size_t)gm * ldc + gn;
                if (C)  C[o]  = val;
                if (Ch) Ch[o] = (_Float16)val;
            }
        }
    }
}

// f32 -> f16 elementwise
__global__ void k_cvt16(const float* __restrict__ X, _Float16* __restrict__ O, long n) {
    long i = (long)blockIdx.x * blockDim.x + threadIdx.x;
    if (i < n) O[i] = (_Float16)X[i];
}

// weight convert + transpose: W[K,N] f32 -> Wt[N,K] f16
__global__ void k_cvt_wt(const float* __restrict__ W, _Float16* __restrict__ Wt, int K, int N) {
    long i = (long)blockIdx.x * blockDim.x + threadIdx.x;
    if (i < (long)K * N) {
        int n = (int)(i / K);
        int k = (int)(i - (long)n * K);
        Wt[i] = (_Float16)W[(size_t)k * N + n];
    }
}

// V [4096,768] f16 -> Vt [96][64][512] f16 (per-head transpose for NT ctx GEMM)
__global__ void k_vtrans(const _Float16* __restrict__ V, _Float16* __restrict__ Vt) {
    long i = (long)blockIdx.x * blockDim.x + threadIdx.x;
    if (i < (long)ZBH * DHd * Sd) {
        int z   = (int)(i >> 15);        // /(64*512)
        long r  = i & 32767;
        int d   = (int)(r >> 9);         // /512
        int kp  = (int)(r & 511);
        int b   = z / NHd;
        int h   = z - b * NHd;
        Vt[i] = V[((size_t)b * Sd + kp) * HIDd + h * DHd + d];
    }
}

// row softmax over 512 keys with additive mask[b,1,1,S]; emits f16 probs
__global__ __launch_bounds__(128) void k_softmax(const float* __restrict__ Sc,
        const float* __restrict__ mask, _Float16* __restrict__ P) {
    const int row = blockIdx.x;               // 0 .. 96*512-1
    const int z   = row >> 9;
    const int b   = z / NHd;
    const float* srow = Sc + (size_t)row * Sd;
    const float* mrow = mask + (size_t)b * Sd;
    __shared__ float red[128];
    const int tid = threadIdx.x;
    float v[4];
    float mx = -3.0e38f;
#pragma unroll
    for (int i = 0; i < 4; ++i) {
        int c = tid + i * 128;
        v[i] = srow[c] + mrow[c];
        mx = fmaxf(mx, v[i]);
    }
    red[tid] = mx; __syncthreads();
    for (int o = 64; o > 0; o >>= 1) { if (tid < o) red[tid] = fmaxf(red[tid], red[tid + o]); __syncthreads(); }
    mx = red[0]; __syncthreads();
    float s = 0.0f;
#pragma unroll
    for (int i = 0; i < 4; ++i) { v[i] = __expf(v[i] - mx); s += v[i]; }
    red[tid] = s; __syncthreads();
    for (int o = 64; o > 0; o >>= 1) { if (tid < o) red[tid] += red[tid + o]; __syncthreads(); }
    const float inv = 1.0f / red[0];
#pragma unroll
    for (int i = 0; i < 4; ++i) {
        int c = tid + i * 128;
        P[(size_t)row * Sd + c] = (_Float16)(v[i] * inv);
    }
}

// out = 0.5*(a+b) as f16 (context averaging)
__global__ void k_avg(const float* __restrict__ A, const float* __restrict__ B,
                      _Float16* __restrict__ O, long n) {
    long i = (long)blockIdx.x * blockDim.x + threadIdx.x;
    if (i < n) O[i] = (_Float16)(0.5f * (A[i] + B[i]));
}

// exact-erf GELU, f16 output
__global__ void k_gelu(const float* __restrict__ H, _Float16* __restrict__ O, long n) {
    long i = (long)blockIdx.x * blockDim.x + threadIdx.x;
    if (i < n) {
        float x = H[i];
        O[i] = (_Float16)(0.5f * x * (1.0f + erff(x * 0.70710678118654752f)));
    }
}

// LayerNorm((y + x)) * g + b over 768 cols; emits f32 and optional f16
__global__ __launch_bounds__(256) void k_add_ln(const float* __restrict__ Y,
        const float* __restrict__ X, const float* __restrict__ g,
        const float* __restrict__ be, float* __restrict__ OF,
        _Float16* __restrict__ OH) {
    const int row = blockIdx.x;
    const int tid = threadIdx.x;
    __shared__ float r1[256], r2[256];
    const float* y = Y + (size_t)row * HIDd;
    const float* x = X + (size_t)row * HIDd;
    float t[3], s1 = 0.0f, s2 = 0.0f;
#pragma unroll
    for (int i = 0; i < 3; ++i) {
        int c = tid + i * 256;
        float u = y[c] + x[c];
        t[i] = u; s1 += u; s2 += u * u;
    }
    r1[tid] = s1; r2[tid] = s2; __syncthreads();
    for (int o = 128; o > 0; o >>= 1) {
        if (tid < o) { r1[tid] += r1[tid + o]; r2[tid] += r2[tid + o]; }
        __syncthreads();
    }
    const float mean = r1[0] * (1.0f / HIDd);
    const float var  = r2[0] * (1.0f / HIDd) - mean * mean;
    const float rstd = rsqrtf(var + 1e-12f);
#pragma unroll
    for (int i = 0; i < 3; ++i) {
        int c = tid + i * 256;
        float o2 = (t[i] - mean) * rstd * g[c] + be[c];
        size_t off = (size_t)row * HIDd + c;
        OF[off] = o2;
        if (OH) OH[off] = (_Float16)o2;
    }
}

// ---------------------------------------------------------------------------
// Host orchestration. Param flattening assumption: jax pytree dict order is
// sorted keys. Top level: audio_cross, audio_ffn, audio_self, lang_cross,
// lang_ffn, lang_self, visn_cross, visn_ffn, visn_self (after 6 feats/masks).
// cross: k1.b,k1.w,k2.b,k2.w,ln.b,ln.g,out.b,out.w,q.b,q.w,v1.b,v1.w,v2.b,v2.w
// self : k.b,k.w,ln.b,ln.g,out.b,out.w,q.b,q.w,v.b,v.w
// ffn  : inter.b,inter.w,ln.b,ln.g,out.b,out.w
// ---------------------------------------------------------------------------
extern "C" void kernel_launch(void* const* d_in, const int* in_sizes, int n_in,
                              void* d_out, int out_size, void* d_ws, size_t ws_size,
                              hipStream_t stream) {
    (void)in_sizes; (void)n_in; (void)out_size; (void)ws_size;

    char* wsb = (char*)d_ws;
    size_t off = 0;
    auto alloc = [&](size_t bytes) -> char* {
        char* p = wsb + off;
        off += (bytes + 255) & ~(size_t)255;
        return p;
    };

    // fixed scratch
    float*     xa[3];  float* xb[3];
    _Float16*  xh[3];  _Float16* xh2[3];
    for (int s = 0; s < 3; ++s) xa[s]  = (float*)alloc((size_t)Md * HIDd * 4);
    for (int s = 0; s < 3; ++s) xb[s]  = (float*)alloc((size_t)Md * HIDd * 4);
    for (int s = 0; s < 3; ++s) xh[s]  = (_Float16*)alloc((size_t)Md * HIDd * 2);
    for (int s = 0; s < 3; ++s) xh2[s] = (_Float16*)alloc((size_t)Md * HIDd * 2);
    _Float16* qbuf  = (_Float16*)alloc((size_t)Md * HIDd * 2);
    _Float16* kbuf  = (_Float16*)alloc((size_t)Md * HIDd * 2);
    _Float16* vbuf  = (_Float16*)alloc((size_t)Md * HIDd * 2);
    _Float16* vt    = (_Float16*)alloc((size_t)ZBH * DHd * Sd * 2);
    float*    score = (float*)alloc((size_t)ZBH * Sd * Sd * 4);
    _Float16* probs = (_Float16*)alloc((size_t)ZBH * Sd * Sd * 2);
    float*    ctx1  = (float*)alloc((size_t)Md * HIDd * 4);
    float*    ctx2  = (float*)alloc((size_t)Md * HIDd * 4);
    _Float16* ctxh  = (_Float16*)alloc((size_t)Md * HIDd * 2);
    float*    yproj = (float*)alloc((size_t)Md * HIDd * 4);
    float*    inter = (float*)alloc((size_t)Md * FFd * 4);
    _Float16* gelh  = (_Float16*)alloc((size_t)Md * FFd * 2);

    auto fptr = [&](int idx) { return (const float*)d_in[idx]; };

    auto cvtW = [&](int idx, int K, int N) -> const _Float16* {
        _Float16* dst = (_Float16*)alloc((size_t)K * N * 2);
        long tot = (long)K * N;
        k_cvt_wt<<<dim3((unsigned)((tot + 255) / 256)), 256, 0, stream>>>(fptr(idx), dst, K, N);
        return dst;
    };

    auto gemm = [&](const _Float16* A, const _Float16* Bt, float* C, _Float16* Ch,
                    const float* bias, float alpha, int M, int N, int K,
                    int lda, int ldb, int ldc, int Z, int nh,
                    long aB, long aH, long bB, long bH, long cB, long cH) {
        if ((M % 128 == 0) && (N % 128 == 0)) {
            dim3 g((unsigned)(M / 128), (unsigned)(N / 128), (unsigned)Z);
            k_gemm_nt<128, 128, 4, 2><<<g, 256, 0, stream>>>(
                A, Bt, C, Ch, bias, alpha, M, N, K, lda, ldb, ldc, nh,
                aB, aH, bB, bH, cB, cH);
        } else {
            dim3 g((unsigned)(M / 64), (unsigned)(N / 64), (unsigned)Z);
            k_gemm_nt<64, 64, 2, 2><<<g, 128, 0, stream>>>(
                A, Bt, C, Ch, bias, alpha, M, N, K, lda, ldb, ldc, nh,
                aB, aH, bB, bH, cB, cH);
        }
    };

    // projection: X[M,K] f16 @ W(idx)[K,N] -> f16 out[M,N]
    auto proj = [&](const _Float16* X, int widx, int bidx, _Float16* Out, int K, int N) {
        const _Float16* Wt = cvtW(widx, K, N);
        gemm(X, Wt, nullptr, Out, fptr(bidx), 1.0f, Md, N, K, K, K, N, 1, 1, 0, 0, 0, 0, 0, 0);
    };

    auto attn = [&](const _Float16* Q, const _Float16* Kc, const _Float16* Vc,
                    const float* mask, float* ctxF) {
        long tv = (long)ZBH * DHd * Sd;
        k_vtrans<<<dim3((unsigned)((tv + 255) / 256)), 256, 0, stream>>>(Vc, vt);
        // scores[z][q][k] = (Q_bh @ K_bh^T)/8 : M=512,N=512,K=64, NT (K matrix is [keys,d])
        gemm(Q, Kc, score, nullptr, nullptr, 0.125f, Sd, Sd, DHd,
             HIDd, HIDd, Sd, ZBH, NHd,
             (long)Sd * HIDd, DHd, (long)Sd * HIDd, DHd,
             (long)NHd * Sd * Sd, (long)Sd * Sd);
        k_softmax<<<dim3(ZBH * Sd), 128, 0, stream>>>(score, mask, probs);
        // ctx[z][q][d] = P @ V : M=512,N=64,K=512 with Vt[z][d][k]
        gemm(probs, vt, ctxF, nullptr, nullptr, 1.0f, Sd, DHd, Sd,
             Sd, Sd, HIDd, ZBH, NHd,
             (long)NHd * Sd * Sd, (long)Sd * Sd, (long)NHd * DHd * Sd, (long)DHd * Sd,
             (long)Sd * HIDd, DHd);
    };

    const float* feats[3] = { fptr(0), fptr(2), fptr(4) };   // lang, visn, audio
    const float* masks[3] = { fptr(1), fptr(3), fptr(5) };
    const int crossB[3] = { 36, 66, 6 };
    const int selfB[3]  = { 56, 86, 26 };
    const int ffnB[3]   = { 50, 80, 20 };
    const int c1s[3] = { 1, 0, 0 }, c2s[3] = { 2, 2, 1 };

    const long nAct = (long)Md * HIDd;
    for (int s = 0; s < 3; ++s)
        k_cvt16<<<dim3((unsigned)((nAct + 255) / 256)), 256, 0, stream>>>(feats[s], xh[s], nAct);

    // ---------------- cross attention ----------------
    for (int s = 0; s < 3; ++s) {
        const int pb = crossB[s], c1 = c1s[s], c2 = c2s[s];
        proj(xh[s],  pb + 9,  pb + 8,  qbuf, HIDd, HIDd);   // q
        proj(xh[c1], pb + 1,  pb + 0,  kbuf, HIDd, HIDd);   // k1
        proj(xh[c1], pb + 11, pb + 10, vbuf, HIDd, HIDd);   // v1
        attn(qbuf, kbuf, vbuf, masks[c1], ctx1);
        proj(xh[c2], pb + 3,  pb + 2,  kbuf, HIDd, HIDd);   // k2
        proj(xh[c2], pb + 13, pb + 12, vbuf, HIDd, HIDd);   // v2
        attn(qbuf, kbuf, vbuf, masks[c2], ctx2);
        k_avg<<<dim3((unsigned)((nAct + 255) / 256)), 256, 0, stream>>>(ctx1, ctx2, ctxh, nAct);
        const _Float16* Wo = cvtW(pb + 7, HIDd, HIDd);
        gemm(ctxh, Wo, yproj, nullptr, fptr(pb + 6), 1.0f, Md, HIDd, HIDd,
             HIDd, HIDd, HIDd, 1, 1, 0, 0, 0, 0, 0, 0);
        k_add_ln<<<dim3(Md), 256, 0, stream>>>(yproj, feats[s], fptr(pb + 5), fptr(pb + 4),
                                               xa[s], xh2[s]);
    }

    // ---------------- self attention ----------------
    for (int s = 0; s < 3; ++s) {
        const int pb = selfB[s];
        proj(xh2[s], pb + 7, pb + 6, qbuf, HIDd, HIDd);     // q
        proj(xh2[s], pb + 1, pb + 0, kbuf, HIDd, HIDd);     // k
        proj(xh2[s], pb + 9, pb + 8, vbuf, HIDd, HIDd);     // v
        attn(qbuf, kbuf, vbuf, masks[s], ctx1);
        k_cvt16<<<dim3((unsigned)((nAct + 255) / 256)), 256, 0, stream>>>(ctx1, ctxh, nAct);
        const _Float16* Wo = cvtW(pb + 5, HIDd, HIDd);
        gemm(ctxh, Wo, yproj, nullptr, fptr(pb + 4), 1.0f, Md, HIDd, HIDd,
             HIDd, HIDd, HIDd, 1, 1, 0, 0, 0, 0, 0, 0);
        k_add_ln<<<dim3(Md), 256, 0, stream>>>(yproj, xa[s], fptr(pb + 3), fptr(pb + 2),
                                               xb[s], xh[s]);
    }

    // ---------------- FFN ----------------
    const long nFF = (long)Md * FFd;
    for (int s = 0; s < 3; ++s) {
        const int pb = ffnB[s];
        const _Float16* Wi = cvtW(pb + 1, HIDd, FFd);
        gemm(xh[s], Wi, inter, nullptr, fptr(pb + 0), 1.0f, Md, FFd, HIDd,
             HIDd, HIDd, FFd, 1, 1, 0, 0, 0, 0, 0, 0);
        k_gelu<<<dim3((unsigned)((nFF + 255) / 256)), 256, 0, stream>>>(inter, gelh, nFF);
        const _Float16* Wo = cvtW(pb + 5, FFd, HIDd);
        gemm(gelh, Wo, yproj, nullptr, fptr(pb + 4), 1.0f, Md, HIDd, FFd,
             FFd, FFd, HIDd, 1, 1, 0, 0, 0, 0, 0, 0);
        float* outp = (float*)d_out + (size_t)s * Md * HIDd;  // stack order lang,visn,audio
        k_add_ln<<<dim3(Md), 256, 0, stream>>>(yproj, xb[s], fptr(pb + 3), fptr(pb + 2),
                                               outp, nullptr);
    }
}
